// ConvStyleGraphConv_24730421691012
// MI455X (gfx1250) — matrix operand order repaired and checked by reading, not compiled, not run on hardware
//
#include <hip/hip_runtime.h>

// ConvStyleGraphConv fused kernel for gfx1250 (MI455X).
// out[b,i,o] = sum_k (sum_j adj_k[i,j] x[b,j,f]) Wcat[k*256+f, o] + bias[o]
//   Phase 0: stage x tile (8 batches) into LDS as bf16
//   Phase A: aggregate z[row=(b,i), ksel*256+f] in LDS via v_fma_mix_f32_bf16
//   Phase B: z @ WT with v_wmma_f32_16x16x32_bf16, f32 accumulate, + bias
// All wave-level guards are routed through readfirstlane -> scalar branches,
// so WMMA never executes under a modified EXEC mask (ISA 7.12 requirement).

#define Jn    17
#define Fin   256
#define Fout  256
#define K3    768
#define BT    8
#define ROWS  (BT * Jn)      // 136 real rows per workgroup
#define ROWSP 144            // padded to 9 M-tiles of 16
#define NMT   9
#define XSTR  264            // bf16 per xs row (256 + 8 pad -> bank-friendly)
#define ZSTR  776            // bf16 per zs row (768 + 8 pad)
#define XS_BYTES (ROWS * XSTR * 2)       // 71,808
#define ZS_BYTES (ROWSP * ZSTR * 2)      // 223,488
#define LDS_BYTES (XS_BYTES + ZS_BYTES + Jn * Jn * 4)   // ~296.5 KB < 320 KB

typedef __attribute__((ext_vector_type(16))) __bf16       v16bf;
typedef __attribute__((ext_vector_type(8)))  __bf16       v8bf;
typedef __attribute__((ext_vector_type(8)))  float        v8f;
typedef __attribute__((ext_vector_type(4)))  float        v4f;
typedef __attribute__((ext_vector_type(4)))  unsigned int v4u;

union ABFrag { v16bf v; v8bf h[2]; };

__device__ __forceinline__ unsigned int pack_bf16(float a, float b) {
  unsigned int ua = __float_as_uint(a);
  unsigned int ub = __float_as_uint(b);
  ua = (ua + 0x7FFFu + ((ua >> 16) & 1u)) >> 16;   // RNE
  ub = (ub + 0x7FFFu + ((ub >> 16) & 1u)) >> 16;
  return ua | (ub << 16);
}

// acc.f32 += bf16(lo/hi half of d) * cf — one VALU op per FMA, no unpack
#define FMA_MIX2(alo, ahi, d, c)                                                  \
  asm("v_fma_mix_f32_bf16 %0, %2, %3, %0 op_sel:[0,0,0] op_sel_hi:[1,0,0]\n\t"    \
      "v_fma_mix_f32_bf16 %1, %2, %3, %1 op_sel:[1,0,0] op_sel_hi:[1,0,0]"        \
      : "+v"(alo), "+v"(ahi)                                                      \
      : "v"(d), "v"(c))

#define WMMA_BF16(ACC, A, Bf)                                                     \
  (ACC) = __builtin_amdgcn_wmma_f32_16x16x32_bf16(false, (A), false, (Bf),        \
                                                  (short)0, (ACC), false, false)

// ---- prep: WT[o][k*256+f] = bf16(W[k][f][o]) : B-operand friendly transpose ----
__global__ void prep_wt(const float* __restrict__ W, unsigned short* __restrict__ WT) {
  int id = blockIdx.x * 256 + threadIdx.x;          // o*768 + kf
  if (id >= Fout * K3) return;
  int o  = id / K3;
  int kf = id - o * K3;
  unsigned int u = __float_as_uint(W[(size_t)kf * Fout + o]);
  u = (u + 0x7FFFu + ((u >> 16) & 1u)) >> 16;
  WT[id] = (unsigned short)u;
}

__global__ __launch_bounds__(512, 1)
void gconv_main(const float* __restrict__ x, const float* __restrict__ adj,
                const float* __restrict__ bias, const __bf16* __restrict__ WT,
                float* __restrict__ out) {
  extern __shared__ char smem[];
  __bf16* xs   = (__bf16*)smem;
  __bf16* zs   = (__bf16*)(smem + XS_BYTES);
  float*  adjs = (float*)(smem + XS_BYTES + ZS_BYTES);

  const int tid  = threadIdx.x;
  const int lane = tid & 31;
  // wave id as a true SGPR value -> all wave-level guards become scalar branches
  const int wu   = __builtin_amdgcn_readfirstlane(tid >> 5);   // 0..15
  const int b0   = blockIdx.x * BT;

  // ---------- Phase 0: stage x -> xs (bf16), adj -> LDS, zero zs pad rows ----------
  {
    const float* xsrc = x + (size_t)b0 * (Jn * Fin);   // 34816 contiguous f32
    unsigned int* xsd = (unsigned int*)xs;
#pragma unroll
    for (int it = 0; it < 17; ++it) {
      int e = (it * 512 + tid) * 4;                    // float index, 16B aligned
      int r = e >> 8;
      int c = e & 255;
      v4f xv = *(const v4f*)(xsrc + e);
      unsigned int* p = xsd + r * (XSTR / 2) + (c >> 1);
      p[0] = pack_bf16(xv.x, xv.y);
      p[1] = pack_bf16(xv.z, xv.w);
    }
    if (tid < Jn * Jn) adjs[tid] = adj[tid];
    // zero pad rows 136..143 of zs (read by WMMA frags, must contribute 0)
    unsigned int* zpad = (unsigned int*)(zs + ROWS * ZSTR);
    for (int i2 = tid; i2 < (ROWSP - ROWS) * (ZSTR / 2); i2 += 512) zpad[i2] = 0u;
  }
  __syncthreads();

  // ---------- Phase A: aggregation. task = (row, ksel), one row per wave ----------
  // z[row][ksel*256+f] = sum_j coef(ksel,i,j) * xs[bi*17+j][f]
  for (int it = 0; it < 26; ++it) {
    int task = wu + (it << 4);                        // scalar, 0..415
    if (task < 3 * ROWS) {
      int ksel = task / ROWS;                         // scalar
      int row  = task - ksel * ROWS;
      int i    = row % Jn;
      int jb, je;
      if (ksel == 0)      { jb = i;     je = i + 1; }
      else if (ksel == 1) { jb = i + 1; je = Jn;    }
      else                { jb = 0;     je = i;     }
      const unsigned int* xp =
          (const unsigned int*)xs + (row - i) * (XSTR / 2) + (lane << 2);
      float a0 = 0.f, a1 = 0.f, a2 = 0.f, a3 = 0.f,
            a4 = 0.f, a5 = 0.f, a6 = 0.f, a7 = 0.f;
      for (int j = jb; j < je; ++j) {
        v4u d = *(const v4u*)(xp + j * (XSTR / 2));
        float cf = adjs[i * Jn + j];
        FMA_MIX2(a0, a1, d.x, cf);
        FMA_MIX2(a2, a3, d.y, cf);
        FMA_MIX2(a4, a5, d.z, cf);
        FMA_MIX2(a6, a7, d.w, cf);
      }
      v4u o4;
      o4.x = pack_bf16(a0, a1); o4.y = pack_bf16(a2, a3);
      o4.z = pack_bf16(a4, a5); o4.w = pack_bf16(a6, a7);
      *(v4u*)((unsigned int*)zs + row * (ZSTR / 2) + ksel * 128 + (lane << 2)) = o4;
    }
  }
  __syncthreads();

  // ---------- Phase B: z @ WT via WMMA bf16 -> f32 ----------
  const int wm = wu >> 2, wn = wu & 3;                // scalar
  const int l15 = lane & 15;
  const int hi16 = (lane & 16) ? 1 : 0;
  const bool extra = (wm == 0);                       // scalar: owns M-tile 8

  float bv[4];
#pragma unroll
  for (int q = 0; q < 4; ++q) bv[q] = bias[(wn * 4 + q) * 16 + l15];

  v8f acc[2][4];                                      // mt = wm, wm+4 (rows 0..127)
#pragma unroll
  for (int p = 0; p < 2; ++p)
#pragma unroll
    for (int q = 0; q < 4; ++q) acc[p][q] = (v8f)0.f;
  v8f acc8[4];                                        // mt = 8 (wm==0 waves only)
#pragma unroll
  for (int q = 0; q < 4; ++q) acc8[q] = (v8f)0.f;

  // per-lane base pointers (ISA 7.12.2 layouts)
  const int a_lo = hi16 ? 8 : 0;    // A: lanes<16 K 0..7/+16..23 ; lanes>=16 +8
  const int b_lo = hi16 ? 8 : 0;    // B: lanes<16 K 0..7/+16..23 ; lanes>=16 +8
  const __bf16* apB[2];
#pragma unroll
  for (int p = 0; p < 2; ++p)
    apB[p] = zs + ((wm + 4 * p) * 16 + l15) * ZSTR + a_lo;
  const __bf16* ap8 = zs + (8 * 16 + l15) * ZSTR + a_lo;
  const __bf16* bpB[4];
#pragma unroll
  for (int q = 0; q < 4; ++q)
    bpB[q] = WT + (size_t)((wn * 4 + q) * 16 + l15) * K3 + b_lo;

#pragma unroll
  for (int kk = 0; kk < 24; ++kk) {
    ABFrag af[2], af8, bf_[4];
#pragma unroll
    for (int p = 0; p < 2; ++p) {
      af[p].h[0] = *(const v8bf*)(apB[p] + kk * 32);
      af[p].h[1] = *(const v8bf*)(apB[p] + kk * 32 + 16);
    }
#pragma unroll
    for (int q = 0; q < 4; ++q) {
      bf_[q].h[0] = *(const v8bf*)(bpB[q] + kk * 32);
      bf_[q].h[1] = *(const v8bf*)(bpB[q] + kk * 32 + 16);
    }
#pragma unroll
    for (int p = 0; p < 2; ++p)
#pragma unroll
      for (int q = 0; q < 4; ++q) WMMA_BF16(acc[p][q], af[p].v, bf_[q].v);
    if (extra) {                                      // scalar branch, EXEC intact
      af8.h[0] = *(const v8bf*)(ap8 + kk * 32);
      af8.h[1] = *(const v8bf*)(ap8 + kk * 32 + 16);
#pragma unroll
      for (int q = 0; q < 4; ++q) WMMA_BF16(acc8[q], af8.v, bf_[q].v);
    }
  }

  // ---------- Epilogue: + bias, store f32 ----------
#pragma unroll
  for (int p = 0; p < 2; ++p) {
    int rlocal = (wm + 4 * p) * 16 + (hi16 ? 8 : 0);  // rows <= 127: all real
#pragma unroll
    for (int q = 0; q < 4; ++q) {
      int col = (wn * 4 + q) * 16 + l15;
      float* po = out + (size_t)(b0 * Jn + rlocal) * Fout + col;
#pragma unroll
      for (int v = 0; v < 8; ++v) po[(size_t)v * Fout] = acc[p][q][v] + bv[q];
    }
  }
  if (extra) {
    // M-tile 8: rows 128..135 real (lanes<16), 136..143 pad (lanes>=16, skip)
    if (!hi16) {
      int rlocal = 128;
#pragma unroll
      for (int q = 0; q < 4; ++q) {
        int col = (wn * 4 + q) * 16 + l15;
        float* po = out + (size_t)(b0 * Jn + rlocal) * Fout + col;
#pragma unroll
        for (int v = 0; v < 8; ++v) po[(size_t)v * Fout] = acc8[q][v] + bv[q];
      }
    }
  }
}

extern "C" void kernel_launch(void* const* d_in, const int* in_sizes, int n_in,
                              void* d_out, int out_size, void* d_ws, size_t ws_size,
                              hipStream_t stream) {
  const float* x    = (const float*)d_in[0];
  const float* W    = (const float*)d_in[1];
  const float* adj  = (const float*)d_in[2];
  const float* bias = (const float*)d_in[3];
  float* out = (float*)d_out;

  unsigned short* WT = (unsigned short*)d_ws;          // 768*256 bf16 = 393,216 B

  prep_wt<<<(Fout * K3 + 255) / 256, 256, 0, stream>>>(W, WT);

  int Btot = in_sizes[0] / (Jn * Fin);                 // 16384
  int nblk = Btot / BT;                                // 2048
  gconv_main<<<nblk, 512, LDS_BYTES, stream>>>(x, adj, bias, (const __bf16*)WT, out);
}
